// ContactMapHead_87548613362504
// MI455X (gfx1250) — compile-verified, matrix operand order checked by reading, not run on hardware
//
#include <hip/hip_runtime.h>
#include <stddef.h>

// ---------------------------------------------------------------------------
// ContactMapHead fused kernel for gfx1250 (MI455X) — bf16x3 split-precision
//   out[b, idx(i,j)] = (X_b @ W @ X_b^T)[l,m] + bias   for valid upper pairs
// Matrix math via V_WMMA_F32_16X16X32_BF16 with hi/lo bf16 splitting:
//   x*y ~= hi_x*hi_y + hi_x*lo_y + lo_x*hi_y  (fp32 accumulate, ~1e-5 rel)
// 3 bf16 WMMAs per K=32 step = 3/8 the matrix-issue count of the f32 path.
// ---------------------------------------------------------------------------

typedef __attribute__((ext_vector_type(8)))  float  v8f;
typedef __attribute__((ext_vector_type(16))) __bf16 v16bf;
typedef __attribute__((ext_vector_type(8)))  __bf16 v8bf;

union Frag16 { v16bf v; v8bf h[2]; };

#define B_ 32
#define L_ 512
#define H_ 1024
#define MT 32            // rows of S per block
#define ROWPB 1032       // LDS row stride in bf16 elems (1024+8: conflict-free)
#define MAXP 130816      // MAX_PAIRS = 511*512/2

// ---- bf16 round-to-nearest-even split helpers (bit math, no __bf16 scalar) -
__device__ __forceinline__ unsigned short bf16_rne(float f) {
    unsigned u = __float_as_uint(f);
    u += 0x7FFFu + ((u >> 16) & 1u);
    return (unsigned short)(u >> 16);
}
__device__ __forceinline__ float bf16_f32(unsigned short h) {
    return __uint_as_float(((unsigned)h) << 16);
}

// ---------------------------------------------------------------------------
// Kernel 0: zero the output (harness poisons d_out with 0xAA)
// ---------------------------------------------------------------------------
__global__ void cmh_zero_kernel(float4* __restrict__ out, size_t n4) {
    size_t i = (size_t)blockIdx.x * blockDim.x + threadIdx.x;
    size_t stride = (size_t)gridDim.x * blockDim.x;
    for (; i < n4; i += stride) out[i] = make_float4(0.f, 0.f, 0.f, 0.f);
}

// ---------------------------------------------------------------------------
// Kernel 1: per-batch mask ranks.
//   r_enc[b,l] = (am==1 && sp==0) ? cumsum-1 : -1 ;  ncount[b] = sum(mask)
// ---------------------------------------------------------------------------
__global__ __launch_bounds__(512)
void cmh_rank_kernel(const int* __restrict__ am, const int* __restrict__ sp,
                     int* __restrict__ r_enc, int* __restrict__ ncount) {
    const int b = blockIdx.x;
    const int t = threadIdx.x;
    __shared__ int s[L_];
    const int m = (am[b * L_ + t] == 1 && sp[b * L_ + t] == 0) ? 1 : 0;
    s[t] = m;
    __syncthreads();
    for (int off = 1; off < L_; off <<= 1) {   // Hillis-Steele inclusive scan
        int v = (t >= off) ? s[t - off] : 0;
        __syncthreads();
        s[t] += v;
        __syncthreads();
    }
    const int incl = s[t];
    r_enc[b * L_ + t] = m ? (incl - 1) : -1;
    if (t == L_ - 1) ncount[b] = incl;
}

// ---------------------------------------------------------------------------
// Kernel 2: split X -> (hi, lo) bf16 arrays (same layout as X)
// ---------------------------------------------------------------------------
__global__ void cmh_splitx_kernel(const float* __restrict__ in,
                                  unsigned short* __restrict__ hi,
                                  unsigned short* __restrict__ lo, size_t n) {
    size_t i = (size_t)blockIdx.x * blockDim.x + threadIdx.x;
    size_t stride = (size_t)gridDim.x * blockDim.x;
    for (; i < n; i += stride) {
        float x = in[i];
        unsigned short h = bf16_rne(x);
        hi[i] = h;
        lo[i] = bf16_rne(x - bf16_f32(h));
    }
}

// ---------------------------------------------------------------------------
// Kernel 3: split + transpose W:  Wt[n][k] = W[k][n]  -> (hi, lo) bf16
// (transposed so WMMA B-fragments are two contiguous 16B loads per lane)
// ---------------------------------------------------------------------------
__global__ void cmh_splitwt_kernel(const float* __restrict__ W,
                                   unsigned short* __restrict__ hi,
                                   unsigned short* __restrict__ lo) {
    int id = blockIdx.x * blockDim.x + threadIdx.x;   // 1M threads exactly
    int n = id >> 10, k = id & (H_ - 1);
    float x = W[k * H_ + n];
    unsigned short h = bf16_rne(x);
    hi[n * H_ + k] = h;
    lo[n * H_ + k] = bf16_rne(x - bf16_f32(h));
}

// ---------------------------------------------------------------------------
// Kernel 4: fused  u = X_tile @ W  (LDS, split bf16)  ->  S = u @ X_b^T
//           -> packed upper-pair scatter, straight from C fragments.
// Block: 256 threads (8 wave32). Grid: (L/MT, B). Dyn LDS: 2*MT*ROWPB*2 B.
// ---------------------------------------------------------------------------
__global__ __launch_bounds__(256, 1)
void cmh_fused_kernel(const unsigned short* __restrict__ Xhi,
                      const unsigned short* __restrict__ Xlo,
                      const unsigned short* __restrict__ Wthi,
                      const unsigned short* __restrict__ Wtlo,
                      const float* __restrict__ bias,
                      const int* __restrict__ r_enc,
                      const int* __restrict__ ncount,
                      float* __restrict__ out) {
    extern __shared__ unsigned short lds[];
    unsigned short* uhi = lds;                    // MT x ROWPB
    unsigned short* ulo = lds + MT * ROWPB;       // MT x ROWPB

    const int batch = blockIdx.y;
    const int I0    = blockIdx.x * MT;
    const int tid   = threadIdx.x;
    const int wave  = tid >> 5;
    const int lane  = tid & 31;
    const int half  = lane >> 4;                  // A/B frag: K-chunk selector
    const int l15   = lane & 15;                  // M (A) / N (B,C)
    const int koff  = half * 8;                   // K = k0+koff+0..7, +16..

    const unsigned short* __restrict__ Xbh = Xhi + (size_t)batch * L_ * H_;
    const unsigned short* __restrict__ Xbl = Xlo + (size_t)batch * L_ * H_;

    // ------------- Stage 1: u[0:32, 0:1024] = X[I0:I0+32, :] @ W -----------
    {
        v8f acc[2][8] = {};     // wave owns 2 M-tiles x 8 N-tiles (nt=wave+8j)
#pragma unroll 1
        for (int k0 = 0; k0 < H_; k0 += 32) {
            Frag16 ah[2], al[2];
#pragma unroll
            for (int mt = 0; mt < 2; ++mt) {
                const size_t ro = (size_t)(I0 + mt * 16 + l15) * H_ + k0 + koff;
                ah[mt].h[0] = *(const v8bf*)(Xbh + ro);
                ah[mt].h[1] = *(const v8bf*)(Xbh + ro + 16);
                al[mt].h[0] = *(const v8bf*)(Xbl + ro);
                al[mt].h[1] = *(const v8bf*)(Xbl + ro + 16);
            }
#pragma unroll
            for (int j = 0; j < 8; ++j) {
                const int nt = wave + 8 * j;
                const size_t co = (size_t)(nt * 16 + l15) * H_ + k0 + koff;
                Frag16 bh, bl;
                bh.h[0] = *(const v8bf*)(Wthi + co);
                bh.h[1] = *(const v8bf*)(Wthi + co + 16);
                bl.h[0] = *(const v8bf*)(Wtlo + co);
                bl.h[1] = *(const v8bf*)(Wtlo + co + 16);
#pragma unroll
                for (int mt = 0; mt < 2; ++mt) {
                    v8f c = acc[mt][j];
                    c = __builtin_amdgcn_wmma_f32_16x16x32_bf16(
                        false, al[mt].v, false, bh.v, (short)0, c, false, false);
                    c = __builtin_amdgcn_wmma_f32_16x16x32_bf16(
                        false, ah[mt].v, false, bl.v, (short)0, c, false, false);
                    c = __builtin_amdgcn_wmma_f32_16x16x32_bf16(
                        false, ah[mt].v, false, bh.v, (short)0, c, false, false);
                    acc[mt][j] = c;
                }
            }
        }
        // split-store u tiles into padded LDS (hi/lo bf16)
#pragma unroll
        for (int mt = 0; mt < 2; ++mt)
#pragma unroll
            for (int j = 0; j < 8; ++j) {
                const int col = (wave + 8 * j) * 16 + l15;
#pragma unroll
                for (int e = 0; e < 8; ++e) {
                    const int M = mt * 16 + e + 8 * half;
                    const float x = acc[mt][j][e];
                    const unsigned short h = bf16_rne(x);
                    uhi[M * ROWPB + col] = h;
                    ulo[M * ROWPB + col] = bf16_rne(x - bf16_f32(h));
                }
            }
    }
    __syncthreads();

    // ------------- Stage 2: S[0:32, 0:512] = u @ X_b^T  + scatter ----------
    {
        v8f acc[2][4] = {};     // wave owns 2 M-tiles x 4 col-tiles (jt=wave+8j)
#pragma unroll 1
        for (int k0 = 0; k0 < H_; k0 += 32) {
            Frag16 ah[2], al[2];
#pragma unroll
            for (int mt = 0; mt < 2; ++mt) {
                const int ro = (mt * 16 + l15) * ROWPB + k0 + koff;
                ah[mt].h[0] = *(const v8bf*)(uhi + ro);      // ds_load_b128
                ah[mt].h[1] = *(const v8bf*)(uhi + ro + 16);
                al[mt].h[0] = *(const v8bf*)(ulo + ro);
                al[mt].h[1] = *(const v8bf*)(ulo + ro + 16);
            }
#pragma unroll
            for (int j = 0; j < 4; ++j) {
                const int jt = wave + 8 * j;
                const size_t co = (size_t)(jt * 16 + l15) * H_ + k0 + koff;
                Frag16 bh, bl;                  // B[k][n] = X_b[n][k]: rows of X
                bh.h[0] = *(const v8bf*)(Xbh + co);
                bh.h[1] = *(const v8bf*)(Xbh + co + 16);
                bl.h[0] = *(const v8bf*)(Xbl + co);
                bl.h[1] = *(const v8bf*)(Xbl + co + 16);
#pragma unroll
                for (int mt = 0; mt < 2; ++mt) {
                    v8f c = acc[mt][j];
                    c = __builtin_amdgcn_wmma_f32_16x16x32_bf16(
                        false, al[mt].v, false, bh.v, (short)0, c, false, false);
                    c = __builtin_amdgcn_wmma_f32_16x16x32_bf16(
                        false, ah[mt].v, false, bl.v, (short)0, c, false, false);
                    c = __builtin_amdgcn_wmma_f32_16x16x32_bf16(
                        false, ah[mt].v, false, bh.v, (short)0, c, false, false);
                    acc[mt][j] = c;
                }
            }
        }

        // fused scatter straight out of the C fragments
        const float bv = bias[0];
        const int   n  = ncount[batch];
        const int* __restrict__ rb = r_enc + batch * L_;
        float* __restrict__ ob = out + (size_t)batch * MAXP;
#pragma unroll
        for (int j = 0; j < 4; ++j) {
            const int gm = (wave + 8 * j) * 16 + l15;     // global column
            const int rj = rb[gm];
#pragma unroll
            for (int mt = 0; mt < 2; ++mt)
#pragma unroll
                for (int e = 0; e < 8; ++e) {
                    const int gl = I0 + mt * 16 + e + 8 * half;   // global row
                    const int ri = rb[gl];
                    if (ri >= 0 && rj >= 0 && gl < gm) {
                        const int idx = ri * (n - 1) - (ri * (ri - 1)) / 2
                                        + (rj - ri - 1);
                        ob[idx] = acc[mt][j][e] + bv;
                    }
                }
        }
    }
}

// ---------------------------------------------------------------------------
// Launch. Inputs (setup_inputs order):
//   0: hidden f32 (32,512,1024)  1: attn i32 (32,512)  2: special i32 (32,512)
//   3: W f32 (1024,1024)         4: b f32 (1,)
// Workspace: ranks 64KB + Xhi/Xlo 2x32MB + Wthi/Wtlo 2x2MB  (~68.1 MB)
// ---------------------------------------------------------------------------
extern "C" void kernel_launch(void* const* d_in, const int* in_sizes, int n_in,
                              void* d_out, int out_size, void* d_ws, size_t ws_size,
                              hipStream_t stream) {
    const float* X    = (const float*)d_in[0];
    const int*   am   = (const int*)d_in[1];
    const int*   sp   = (const int*)d_in[2];
    const float* W    = (const float*)d_in[3];
    const float* bias = (const float*)d_in[4];
    float*       out  = (float*)d_out;

    char* ws = (char*)d_ws;
    int* r_enc  = (int*)ws;                               // 32*512 ints
    int* ncount = (int*)(ws + B_ * L_ * sizeof(int));     // 32 ints
    const size_t xele = (size_t)B_ * L_ * H_;             // 16,777,216
    unsigned short* Xhi  = (unsigned short*)(ws + 66048); // 256B-aligned region
    unsigned short* Xlo  = Xhi + xele;
    unsigned short* Wthi = Xlo + xele;
    unsigned short* Wtlo = Wthi + (size_t)H_ * H_;

    // 1) zero output (32*130816 floats, divisible by 4)
    const size_t n4 = (size_t)B_ * MAXP / 4;
    cmh_zero_kernel<<<4096, 256, 0, stream>>>((float4*)out, n4);

    // 2) mask ranks
    cmh_rank_kernel<<<B_, L_, 0, stream>>>(am, sp, r_enc, ncount);

    // 3) split X into bf16 hi/lo
    cmh_splitx_kernel<<<8192, 256, 0, stream>>>(X, Xhi, Xlo, xele);

    // 4) split + transpose W
    cmh_splitwt_kernel<<<(H_ * H_) / 256, 256, 0, stream>>>(W, Wthi, Wtlo);

    // 5) fused GEMM+GEMM+scatter (bf16x3 WMMA)
    dim3 grid(L_ / MT, B_);
    const size_t lds_bytes = 2u * MT * ROWPB * sizeof(unsigned short); // 132096
    cmh_fused_kernel<<<grid, 256, lds_bytes, stream>>>(Xhi, Xlo, Wthi, Wtlo,
                                                       bias, r_enc, ncount, out);
}
